// TransformerEncoder_51204600103624
// MI455X (gfx1250) — matrix operand order repaired and checked by reading, not compiled, not run on hardware
//
#include <hip/hip_runtime.h>
#include <hip/hip_bf16.h>
#include <math.h>

// ---------------------------------------------------------------------------
// Types for CDNA5 WMMA (wave32, 16x16x32 bf16 -> f32) and TDM descriptors
// ---------------------------------------------------------------------------
typedef __bf16 bf16_t;
typedef __attribute__((ext_vector_type(16))) __bf16 v16bf;
typedef __attribute__((ext_vector_type(8)))  __bf16 v8bf;
typedef __attribute__((ext_vector_type(8)))  float  v8f;
typedef __attribute__((ext_vector_type(4)))  unsigned int u32x4;
typedef __attribute__((ext_vector_type(4)))  int i32x4;
typedef __attribute__((ext_vector_type(8)))  int i32x8;

union FragBF { v16bf v; v8bf h[2]; };

__device__ __forceinline__ bf16_t f2bf(float f) {
  unsigned u = __float_as_uint(f);
  u += 0x7FFFu + ((u >> 16) & 1u);          // round-to-nearest-even
  unsigned short hs = (unsigned short)(u >> 16);
  union { unsigned short s; bf16_t b; } o; o.s = hs; return o.b;
}

__device__ __forceinline__ float gelu_exact(float x) {
  return 0.5f * x * (1.0f + erff(x * 0.7071067811865475f));
}

__device__ __forceinline__ unsigned fkey(float f) {
  unsigned u = __float_as_uint(f);
  return (u & 0x80000000u) ? ~u : (u | 0x80000000u);
}
__device__ __forceinline__ float funkey(unsigned k) {
  unsigned u = (k & 0x80000000u) ? (k & 0x7FFFFFFFu) : ~k;
  return __uint_as_float(u);
}

// ---------------------------------------------------------------------------
// Tensor Data Mover: 2D bf16 tile (tile1 rows x tile0 elems) -> LDS.
// D# built per CDNA5 ISA 8.3/8.4: group0 {count, lds_addr, global_addr, type=2},
// group1 {data_size=1(2B), tensor_dim0/1 = remaining extent from tile start,
//         tile_dim0/1, tensor_dim0_stride}.  Issued by one wave; TENSORcnt.
// ---------------------------------------------------------------------------
__device__ __forceinline__ void tdm_load_2d(void* lds, const void* gptr,
                                            int stride_elems, int d0, int d1,
                                            int tile0, int tile1)
{
  unsigned long long ga = (unsigned long long)(size_t)gptr;
  unsigned ldsa = (unsigned)(size_t)lds;     // low 32 bits of flat LDS addr
  u32x4 g0;
  g0[0] = 1u;                                 // count=1, user descriptor
  g0[1] = ldsa;                               // lds_addr
  g0[2] = (unsigned)(ga & 0xFFFFFFFFu);       // global_addr[31:0]
  g0[3] = (unsigned)((ga >> 32) & 0x01FFFFFFu) | (2u << 30);  // [56:32] | type=2
  i32x8 g1;
  g1[0] = 0x00010000;                         // data_size=1 (2 bytes)
  g1[1] = (int)(((unsigned)d0 & 0xFFFFu) << 16);              // tensor_dim0 lo
  g1[2] = (int)((((unsigned)d0 >> 16) & 0xFFFFu) |
                (((unsigned)d1 & 0xFFFFu) << 16));            // dim0 hi | dim1 lo
  g1[3] = (int)((((unsigned)d1 >> 16) & 0xFFFFu) |
                ((unsigned)tile0 << 16));                     // dim1 hi | tile0
  g1[4] = tile1;                              // tile_dim1 | tile_dim2=0
  g1[5] = stride_elems;                       // tensor_dim0_stride[31:0]
  g1[6] = 0;
  g1[7] = 0;
  i32x4 gz = {0, 0, 0, 0};
#if defined(__clang_major__) && (__clang_major__ >= 23)
  i32x8 gz8 = {0, 0, 0, 0, 0, 0, 0, 0};
  __builtin_amdgcn_tensor_load_to_lds(g0, g1, gz, gz, gz8, 0);
#else
  __builtin_amdgcn_tensor_load_to_lds(g0, g1, gz, gz, 0);
#endif
}

// ---------------------------------------------------------------------------
// bf16 WMMA GEMM with TDM tile fill:
//   C = act( A(MxK bf16, lda) * Bt(NxK bf16, ldb)^T + bias + resid )
// Tiles: 128x128 output, TK=64; 256 threads = 8 waves, wave = 32x64 (2x4 frags).
// Both A and Bt tiles are row-major 2D rectangles -> one TDM descriptor each,
// double-buffered, synced with s_wait_tensorcnt + barrier.
// ---------------------------------------------------------------------------
#define TM 128
#define TN 128
#define TK 64

__global__ __launch_bounds__(256)
void gemm_bf16_wmma(const bf16_t* __restrict__ A, int lda,
                    const bf16_t* __restrict__ Bt, int ldb,
                    float* __restrict__ C, int ldc,
                    const float* __restrict__ bias,
                    const float* __restrict__ resid, int ldr,
                    int M, int N, int K, int act)
{
  __shared__ bf16_t As[2][TM][TK];   // 16 KB each buffer
  __shared__ bf16_t Bs[2][TN][TK];   // [n][k], 16 KB each buffer

  const int tid  = threadIdx.x;
  const int lane = tid & 31;
  const int wave = tid >> 5;
  const int half = lane >> 4;
  const int l16  = lane & 15;
  const int bm = blockIdx.x * TM;
  const int bn = blockIdx.y * TN;
  const int wm = (wave >> 1) * 32;
  const int wn = (wave & 1) * 64;

  v8f acc[2][4];
  #pragma unroll
  for (int i = 0; i < 2; ++i)
    #pragma unroll
    for (int j = 0; j < 4; ++j)
      #pragma unroll
      for (int r = 0; r < 8; ++r) acc[i][j][r] = 0.0f;

  const int ksteps = K / TK;

  if (tid < 32) {   // wave 0 drives the TDM
    tdm_load_2d(&As[0][0][0], A + (size_t)bm * lda, lda, K, M - bm, TK, TM);
    tdm_load_2d(&Bs[0][0][0], Bt + (size_t)bn * ldb, ldb, K, N - bn, TK, TN);
  }

  for (int kt = 0; kt < ksteps; ++kt) {
    const int cur = kt & 1;
    if (tid < 32) {
      if (kt + 1 < ksteps) {
        int k0 = (kt + 1) * TK;
        tdm_load_2d(&As[cur ^ 1][0][0], A + (size_t)bm * lda + k0, lda,
                    K - k0, M - bm, TK, TM);
        tdm_load_2d(&Bs[cur ^ 1][0][0], Bt + (size_t)bn * ldb + k0, ldb,
                    K - k0, N - bn, TK, TN);
        __builtin_amdgcn_s_wait_tensorcnt(2);   // tile kt done, kt+1 in flight
      } else {
        __builtin_amdgcn_s_wait_tensorcnt(0);
      }
    }
    __syncthreads();                            // tile kt visible to all waves

    #pragma unroll
    for (int s = 0; s < 2; ++s) {               // two k=32 sub-steps
      FragBF afr[2], bfr[4];
      #pragma unroll
      for (int i = 0; i < 2; ++i) {
        const bf16_t* ap = &As[cur][wm + i * 16 + l16][s * 32];
        afr[i].h[0] = *(const v8bf*)(ap + half * 8);
        afr[i].h[1] = *(const v8bf*)(ap + 16 + half * 8);
      }
      #pragma unroll
      for (int j = 0; j < 4; ++j) {
        const bf16_t* bp = &Bs[cur][wn + j * 16 + l16][s * 32];
        bfr[j].h[0] = *(const v8bf*)(bp + half * 8);
        bfr[j].h[1] = *(const v8bf*)(bp + 16 + half * 8);
      }
      #pragma unroll
      for (int i = 0; i < 2; ++i)
        #pragma unroll
        for (int j = 0; j < 4; ++j)
          acc[i][j] = __builtin_amdgcn_wmma_f32_16x16x32_bf16(
              false, afr[i].v, false, bfr[j].v, (short)0, acc[i][j],
              false, false);
    }
    __syncthreads();                            // reads done before overwrite
  }

  #pragma unroll
  for (int i = 0; i < 2; ++i) {
    #pragma unroll
    for (int j = 0; j < 4; ++j) {
      int col = bn + wn + j * 16 + l16;
      if (col >= N) continue;
      float bv = bias ? bias[col] : 0.0f;
      #pragma unroll
      for (int r = 0; r < 8; ++r) {
        int row = bm + wm + i * 16 + half * 8 + r;
        if (row >= M) continue;
        float v = acc[i][j][r] + bv;
        if (resid) v += resid[(size_t)row * ldr + col];
        if (act) v = gelu_exact(v);
        C[(size_t)row * ldc + col] = v;
      }
    }
  }
}

// ---------------------------------------------------------------------------
// Prep: f32 -> bf16 convert (grid-stride over float4) and
//       f32 (KxN) -> bf16 transposed (NxK) weight convert.
// ---------------------------------------------------------------------------
__global__ __launch_bounds__(256)
void cvt_bf16(const float* __restrict__ in, bf16_t* __restrict__ out, long n4)
{
  long i = (long)blockIdx.x * 256 + threadIdx.x;
  const long stride = (long)gridDim.x * 256;
  for (; i < n4; i += stride) {
    float4 v = ((const float4*)in)[i];
    bf16_t* o = out + i * 4;
    o[0] = f2bf(v.x); o[1] = f2bf(v.y); o[2] = f2bf(v.z); o[3] = f2bf(v.w);
  }
}

__global__ __launch_bounds__(256)
void tr_bf16(const float* __restrict__ in, bf16_t* __restrict__ out,
             int K, int N)
{
  __shared__ float t[32][33];
  int k0 = blockIdx.x * 32, n0 = blockIdx.y * 32;
  int tx = threadIdx.x, ty = threadIdx.y;     // (32, 8)
  #pragma unroll
  for (int j = 0; j < 4; ++j) {
    int k = k0 + ty + j * 8;
    t[ty + j * 8][tx] = (k < K && n0 + tx < N) ? in[(size_t)k * N + n0 + tx]
                                               : 0.0f;
  }
  __syncthreads();
  #pragma unroll
  for (int j = 0; j < 4; ++j) {
    int n = n0 + ty + j * 8;
    if (n < N && k0 + tx < K)
      out[(size_t)n * K + k0 + tx] = f2bf(t[tx][ty + j * 8]);
  }
}

// ---------------------------------------------------------------------------
// Attention (q/k/v/o as (B, m=4, D=1024) f32; head h = cols h*64..h*64+63)
// ---------------------------------------------------------------------------
__global__ __launch_bounds__(256)
void attn_scores(const float* __restrict__ q, const float* __restrict__ k,
                 float* __restrict__ s, unsigned* __restrict__ gminkey)
{
  int wave = threadIdx.x >> 5, lane = threadIdx.x & 31;
  int bh = blockIdx.x * 8 + wave;
  int b = bh >> 4, h = bh & 15;
  const float* qb = q + (size_t)b * 4096 + h * 64;
  const float* kb = k + (size_t)b * 4096 + h * 64;
  float sc = 1e30f;
  if (lane < 16) {
    int r = lane >> 2, c = lane & 3;
    const float4* qp = (const float4*)(qb + r * 1024);
    const float4* kp = (const float4*)(kb + c * 1024);
    float a = 0.f;
    #pragma unroll
    for (int i = 0; i < 16; ++i) {
      float4 x = qp[i], y = kp[i];
      a += x.x * y.x + x.y * y.y + x.z * y.z + x.w * y.w;
    }
    sc = a * 0.125f;
    s[(size_t)bh * 16 + lane] = sc;
  }
  float m = sc;
  #pragma unroll
  for (int o = 16; o >= 1; o >>= 1) m = fminf(m, __shfl_xor(m, o));
  if (lane == 0) atomicMin(gminkey, fkey(m));
}

__global__ __launch_bounds__(256)
void attn_out(const float* __restrict__ s, const unsigned* __restrict__ gminkey,
              const float* __restrict__ v, float* __restrict__ o)
{
  __shared__ float ps[8][16];
  int wave = threadIdx.x >> 5, lane = threadIdx.x & 31;
  int bh = blockIdx.x * 8 + wave;
  int b = bh >> 4, h = bh & 15;
  float inv = 1.0f / fmaxf(fabsf(funkey(*gminkey)), 1e-30f);
  if (lane < 16) {
    float a = s[(size_t)bh * 16 + lane] * inv;
    float sq = a * a;
    sq += __shfl_xor(sq, 1); sq += __shfl_xor(sq, 2);
    a = a / fmaxf(sqrtf(sq), 1e-12f);
    float mx = a;
    mx = fmaxf(mx, __shfl_xor(mx, 1)); mx = fmaxf(mx, __shfl_xor(mx, 2));
    float e = __expf(a - mx);
    float se = e; se += __shfl_xor(se, 1); se += __shfl_xor(se, 2);
    ps[wave][lane] = e / se;
  }
  __syncthreads();
  const float* vb = v + (size_t)b * 4096 + h * 64;
  float*       ob = o + (size_t)b * 4096 + h * 64;
  int d = lane * 2;
  #pragma unroll
  for (int r = 0; r < 4; ++r) {
    float2 accv = make_float2(0.f, 0.f);
    #pragma unroll
    for (int c = 0; c < 4; ++c) {
      float p = ps[wave][r * 4 + c];
      float2 vv = *(const float2*)(vb + c * 1024 + d);
      accv.x += p * vv.x; accv.y += p * vv.y;
    }
    *(float2*)(ob + r * 1024 + d) = accv;
  }
}

// ---------------------------------------------------------------------------
// LayerNorm over rows of length n; one 256-thread block per row.
// ---------------------------------------------------------------------------
__global__ __launch_bounds__(256)
void ln_rows(const float* __restrict__ in, float* __restrict__ out,
             const float* __restrict__ g, const float* __restrict__ b,
             int n, float eps)
{
  __shared__ float red[8];
  __shared__ float stat[2];
  size_t row = blockIdx.x;
  const float* x = in + row * (size_t)n;
  float*       y = out + row * (size_t)n;
  int wave = threadIdx.x >> 5, lane = threadIdx.x & 31;

  float sm = 0.f;
  for (int i = threadIdx.x; i < n; i += 256) sm += x[i];
  #pragma unroll
  for (int o = 16; o >= 1; o >>= 1) sm += __shfl_xor(sm, o);
  if (lane == 0) red[wave] = sm;
  __syncthreads();
  if (threadIdx.x == 0) {
    float t = 0.f;
    for (int i = 0; i < 8; ++i) t += red[i];
    stat[0] = t / (float)n;
  }
  __syncthreads();
  float mean = stat[0];

  float vv = 0.f;
  for (int i = threadIdx.x; i < n; i += 256) { float d = x[i] - mean; vv += d * d; }
  #pragma unroll
  for (int o = 16; o >= 1; o >>= 1) vv += __shfl_xor(vv, o);
  if (lane == 0) red[wave] = vv;
  __syncthreads();
  if (threadIdx.x == 0) {
    float t = 0.f;
    for (int i = 0; i < 8; ++i) t += red[i];
    stat[1] = 1.0f / sqrtf(t / (float)n + eps);
  }
  __syncthreads();
  float invs = stat[1];
  for (int i = threadIdx.x; i < n; i += 256)
    y[i] = (x[i] - mean) * invs * g[i] + b[i];
}

__global__ void init_keys(unsigned* g) {
  if (threadIdx.x < 8) g[threadIdx.x] = 0xFFFFFFFFu;
}

// ---------------------------------------------------------------------------
// Host-side pipeline
// ---------------------------------------------------------------------------
static inline void launch_gemm(hipStream_t st,
                               const bf16_t* A, int lda, const bf16_t* Bt, int ldb,
                               float* C, int ldc, const float* bias,
                               const float* resid, int ldr,
                               int M, int N, int K, int act)
{
  dim3 grid(M / 128, (N + 127) / 128), blk(256);
  gemm_bf16_wmma<<<grid, blk, 0, st>>>(A, lda, Bt, ldb, C, ldc, bias, resid, ldr,
                                       M, N, K, act);
}

static inline void launch_tr(hipStream_t st, const float* in, bf16_t* out,
                             int K, int N)
{
  dim3 grid((K + 31) / 32, (N + 31) / 32), blk(32, 8);
  tr_bf16<<<grid, blk, 0, st>>>(in, out, K, N);
}

static inline void launch_cvt(hipStream_t st, const float* in, bf16_t* out,
                              size_t n)
{
  cvt_bf16<<<2048, 256, 0, st>>>(in, out, (long)(n / 4));
}

extern "C" void kernel_launch(void* const* d_in, const int* in_sizes, int n_in,
                              void* d_out, int out_size, void* d_ws, size_t ws_size,
                              hipStream_t stream)
{
  (void)in_sizes; (void)n_in; (void)out_size; (void)ws_size;
  const int Bsz = 4096, D = 1024, Mrows = 16384;
  const int dims[4] = {1024, 2048, 4096, 8192};
  const int offs[4] = {0, 1024, 3072, 7168};
  const int DD = D * D;

  const float* x = (const float*)d_in[0];
  const float* wqm[4] = {(const float*)d_in[1], (const float*)d_in[4],
                         (const float*)d_in[7], (const float*)d_in[10]};
  const float* wkm[4] = {(const float*)d_in[2], (const float*)d_in[5],
                         (const float*)d_in[8], (const float*)d_in[11]};
  const float* wvm[4] = {(const float*)d_in[3], (const float*)d_in[6],
                         (const float*)d_in[9], (const float*)d_in[12]};
  const float* fc_in_w = (const float*)d_in[13];
  const float* fc_in_b = (const float*)d_in[14];
  const float* ln_in_g = (const float*)d_in[15];
  const float* ln_in_b = (const float*)d_in[16];
  const float* Wq = (const float*)d_in[17];
  const float* Wk = (const float*)d_in[18];
  const float* Wv = (const float*)d_in[19];
  const float* Fc = (const float*)d_in[20];
  const float* W1 = (const float*)d_in[21];
  const float* W2 = (const float*)d_in[22];
  const float* B1 = (const float*)d_in[23];
  const float* B2 = (const float*)d_in[24];
  const float* L1g = (const float*)d_in[25];
  const float* L1b = (const float*)d_in[26];
  const float* L2g = (const float*)d_in[27];
  const float* L2b = (const float*)d_in[28];
  const float* OLg = (const float*)d_in[29];
  const float* OLb = (const float*)d_in[30];
  const float* OW1 = (const float*)d_in[31];
  const float* OB1 = (const float*)d_in[32];
  const float* OW2 = (const float*)d_in[33];
  const float* OB2 = (const float*)d_in[34];
  float* out = (float*)d_out;

  // ---- workspace carve-up -------------------------------------------------
  char* w = (char*)d_ws;
  auto alloc_f = [&](size_t n) { float* p = (float*)w; w += n * 4; return p; };
  auto alloc_h = [&](size_t n) { bf16_t* p = (bf16_t*)w; w += ((n + 7) & ~7ull) * 2; return p; };

  const size_t S = (size_t)Mrows * D;               // 16.78M elems
  float* qbuf = alloc_f(S);
  float* kbuf = alloc_f(S);
  float* vbuf = alloc_f(S);
  float* obuf = alloc_f(S);
  float* hbuf = alloc_f(S);
  float* tbuf = alloc_f(S);
  float* fbuf = alloc_f(S);
  float* sbuf = alloc_f((size_t)65536 * 16);
  unsigned* gmin = (unsigned*)alloc_f(8);

  bf16_t* xbf = alloc_h((size_t)Bsz * 15360);
  bf16_t* abf = alloc_h(S);                          // bf16 activation staging
  bf16_t *wqT[4], *wkT[4], *wvT[4];
  for (int i = 0; i < 4; ++i) {
    wqT[i] = alloc_h((size_t)D * dims[i]);
    wkT[i] = alloc_h((size_t)D * dims[i]);
    wvT[i] = alloc_h((size_t)D * dims[i]);
  }
  bf16_t* fcinT = alloc_h(DD);
  bf16_t* WqT = alloc_h((size_t)4 * DD);
  bf16_t* WkT = alloc_h((size_t)4 * DD);
  bf16_t* WvT = alloc_h((size_t)4 * DD);
  bf16_t* FcT = alloc_h((size_t)4 * DD);
  bf16_t* W1T = alloc_h((size_t)4 * DD);
  bf16_t* W2T = alloc_h((size_t)4 * DD);
  bf16_t* ow1T = alloc_h((size_t)D * 4096);
  bf16_t* ow2T = alloc_h((size_t)10 * D);

  // ---- prep: converts / weight transposes ---------------------------------
  init_keys<<<1, 32, 0, stream>>>(gmin);
  launch_cvt(stream, x, xbf, (size_t)Bsz * 15360);
  for (int i = 0; i < 4; ++i) {
    launch_tr(stream, wqm[i], wqT[i], dims[i], D);
    launch_tr(stream, wkm[i], wkT[i], dims[i], D);
    launch_tr(stream, wvm[i], wvT[i], dims[i], D);
  }
  launch_tr(stream, fc_in_w, fcinT, D, D);
  for (int l = 0; l < 4; ++l) {
    launch_tr(stream, Wq + (size_t)l * DD, WqT + (size_t)l * DD, D, D);
    launch_tr(stream, Wk + (size_t)l * DD, WkT + (size_t)l * DD, D, D);
    launch_tr(stream, Wv + (size_t)l * DD, WvT + (size_t)l * DD, D, D);
    launch_tr(stream, Fc + (size_t)l * DD, FcT + (size_t)l * DD, D, D);
    launch_tr(stream, W1 + (size_t)l * DD, W1T + (size_t)l * DD, D, D);
    launch_tr(stream, W2 + (size_t)l * DD, W2T + (size_t)l * DD, D, D);
  }
  launch_tr(stream, OW1, ow1T, 4096, D);
  launch_tr(stream, OW2, ow2T, D, 10);

  // ---- modality QKV projections -> (B, m, D) layout -----------------------
  for (int i = 0; i < 4; ++i) {
    launch_gemm(stream, xbf + offs[i], 15360, wqT[i], dims[i], qbuf + i * D,
                4 * D, nullptr, nullptr, 0, Bsz, D, dims[i], 0);
    launch_gemm(stream, xbf + offs[i], 15360, wkT[i], dims[i], kbuf + i * D,
                4 * D, nullptr, nullptr, 0, Bsz, D, dims[i], 0);
    launch_gemm(stream, xbf + offs[i], 15360, wvT[i], dims[i], vbuf + i * D,
                4 * D, nullptr, nullptr, 0, Bsz, D, dims[i], 0);
  }

  // ---- input attention + fc_in + residual(v) + LN -------------------------
  attn_scores<<<8192, 256, 0, stream>>>(qbuf, kbuf, sbuf, gmin + 0);
  attn_out<<<8192, 256, 0, stream>>>(sbuf, gmin + 0, vbuf, obuf);
  launch_cvt(stream, obuf, abf, S);
  launch_gemm(stream, abf, D, fcinT, D, tbuf, D, fc_in_b, vbuf, D,
              Mrows, D, D, 0);
  ln_rows<<<Mrows, 256, 0, stream>>>(tbuf, hbuf, ln_in_g, ln_in_b, D, 1e-6f);

  // ---- encoder layers -----------------------------------------------------
  for (int l = 0; l < 4; ++l) {
    launch_cvt(stream, hbuf, abf, S);
    launch_gemm(stream, abf, D, WqT + (size_t)l * DD, D, qbuf, D,
                nullptr, nullptr, 0, Mrows, D, D, 0);
    launch_gemm(stream, abf, D, WkT + (size_t)l * DD, D, kbuf, D,
                nullptr, nullptr, 0, Mrows, D, D, 0);
    launch_gemm(stream, abf, D, WvT + (size_t)l * DD, D, vbuf, D,
                nullptr, nullptr, 0, Mrows, D, D, 0);
    attn_scores<<<8192, 256, 0, stream>>>(qbuf, kbuf, sbuf, gmin + 1 + l);
    attn_out<<<8192, 256, 0, stream>>>(sbuf, gmin + 1 + l, vbuf, obuf);
    launch_cvt(stream, obuf, abf, S);
    launch_gemm(stream, abf, D, FcT + (size_t)l * DD, D, tbuf, D,
                nullptr, hbuf, D, Mrows, D, D, 0);
    ln_rows<<<Mrows, 256, 0, stream>>>(tbuf, hbuf, L1g + l * D, L1b + l * D,
                                       D, 1e-6f);
    launch_cvt(stream, hbuf, abf, S);
    launch_gemm(stream, abf, D, W1T + (size_t)l * DD, D, fbuf, D,
                B1 + l * D, nullptr, 0, Mrows, D, D, 1 /*gelu*/);
    launch_cvt(stream, fbuf, abf, S);
    launch_gemm(stream, abf, D, W2T + (size_t)l * DD, D, tbuf, D,
                B2 + l * D, hbuf, D, Mrows, D, D, 0);
    ln_rows<<<Mrows, 256, 0, stream>>>(tbuf, hbuf, L2g + l * D, L2b + l * D,
                                       D, 1e-6f);
  }

  // ---- output head --------------------------------------------------------
  float* zbuf  = qbuf;   // (4096 x 4096) f32, q/k dead now
  float* z1buf = kbuf;   // (4096 x 1024) f32
  ln_rows<<<Bsz, 256, 0, stream>>>(hbuf, zbuf, OLg, OLb, 4 * D, 1e-5f);
  launch_cvt(stream, zbuf, abf, S);
  launch_gemm(stream, abf, 4 * D, ow1T, 4 * D, z1buf, D, OB1, nullptr, 0,
              Bsz, D, 4 * D, 1 /*gelu*/);
  launch_cvt(stream, z1buf, abf, (size_t)Bsz * D);
  launch_gemm(stream, abf, D, ow2T, D, out, 10, OB2, nullptr, 0,
              Bsz, 10, D, 0);
}
